// AdaptiveSliceSelector_79242146611958
// MI455X (gfx1250) — compile-verified, matrix operand order checked by reading, not run on hardware
//
#include <hip/hip_runtime.h>

#define H 256
#define TM 128      // block tile M
#define TNB 64      // block tile N
#define KC 32       // K chunk (one WMMA depth)
#define APAD 40     // padded row length (80B: 16B-aligned, conflict-free for b128)

typedef __bf16 bf16;
typedef __attribute__((ext_vector_type(16))) __bf16 v16bf;
typedef __attribute__((ext_vector_type(8)))  __bf16 v8bf;
typedef __attribute__((ext_vector_type(8)))  float  v8f;

// ---------- helpers ----------
__device__ __forceinline__ unsigned short f2bfbits(float f) {
  unsigned u = __float_as_uint(f);
  return (unsigned short)((u + 0x7FFFu + ((u >> 16) & 1u)) >> 16);  // RNE
}
__device__ __forceinline__ float bfbits2f(unsigned short h) {
  return __uint_as_float(((unsigned)h) << 16);
}
__device__ __forceinline__ bf16 f2bf(float f) {
  unsigned short hs = f2bfbits(f);
  return __builtin_bit_cast(bf16, hs);
}
__device__ __forceinline__ float lrelu(float v) { return v > 0.f ? v : 0.2f * v; }
// order-preserving float<->uint map for atomic max on floats
__device__ __forceinline__ unsigned f2ord(float f) {
  unsigned u = __float_as_uint(f);
  return (u & 0x80000000u) ? ~u : (u | 0x80000000u);
}
__device__ __forceinline__ float ord2f(unsigned u) {
  unsigned b = (u & 0x80000000u) ? (u & 0x7FFFFFFFu) : ~u;
  return __uint_as_float(b);
}

// fragment: 16 bf16 per lane = two 16B LDS vector loads (conflict-free via APAD)
__device__ __forceinline__ v16bf load_frag(const bf16* rowp, int g0) {
  v8bf a = *(const v8bf*)(rowp + g0);        // K = g0..g0+7
  v8bf b = *(const v8bf*)(rowp + 16 + g0);   // K = 16+g0..16+g0+7
  return __builtin_shufflevector(a, b, 0, 1, 2, 3, 4, 5, 6, 7,
                                 8, 9, 10, 11, 12, 13, 14, 15);
}

#define WMMA_BF16(A_, B_, C_) \
  __builtin_amdgcn_wmma_f32_16x16x32_bf16(false, (A_), false, (B_), (short)0, (C_), false, false)

// ---------- bf16x3 split-precision WMMA GEMM:  C[M,256] = A[M,256] @ B[256,256] (+bias) ----------
__global__ __launch_bounds__(256) void k_gemm_bf16x3(
    const float* __restrict__ A, const float* __restrict__ B,
    const float* __restrict__ bias, float* __restrict__ C, int M)
{
  __shared__ bf16 Ah[TM][APAD];
  __shared__ bf16 Al[TM][APAD];
  __shared__ bf16 Bth[TNB][APAD];   // B transposed: [col][K]
  __shared__ bf16 Btl[TNB][APAD];

  const int tid = threadIdx.x;
  const int m0 = blockIdx.x * TM;
  const int n0 = blockIdx.y * TNB;
  const int wid = tid >> 5, lane = tid & 31;
  const int hl = lane & 15, hh = lane >> 4;
  const int wr = wid >> 1;                  // 0..3 -> M offset wr*32
  const int wc = wid & 1;                   // 0..1 -> N offset wc*32
  const int g0 = hh << 3;

  v8f acc00 = {}, acc01 = {}, acc10 = {}, acc11 = {};

  for (int k0 = 0; k0 < H; k0 += KC) {
    __syncthreads();
    // ---- stage A tile: 128x32 f32, float4 loads, b64 LDS stores of 4 packed bf16
    for (int idx = tid; idx < (TM * KC) / 4; idx += 256) {   // 1024 -> 4 iters
      int r = idx >> 3;                 // 8 float4 per row
      int c4 = (idx & 7) << 2;
      int gr = m0 + r;
      float4 v = (gr < M) ? *(const float4*)&A[(size_t)gr * H + k0 + c4]
                          : make_float4(0.f, 0.f, 0.f, 0.f);
      ushort4 hiv, lov;
      hiv.x = f2bfbits(v.x); lov.x = f2bfbits(v.x - bfbits2f(hiv.x));
      hiv.y = f2bfbits(v.y); lov.y = f2bfbits(v.y - bfbits2f(hiv.y));
      hiv.z = f2bfbits(v.z); lov.z = f2bfbits(v.z - bfbits2f(hiv.z));
      hiv.w = f2bfbits(v.w); lov.w = f2bfbits(v.w - bfbits2f(hiv.w));
      *(ushort4*)&Ah[r][c4] = hiv;
      *(ushort4*)&Al[r][c4] = lov;
    }
    // ---- stage B tile transposed: B is [K][256] row-major -> Bt[col][K]
    for (int idx = tid; idx < (KC * TNB) / 4; idx += 256) {  // 512 -> 2 iters
      int r = idx >> 4;                 // 16 float4 per K row
      int c4 = (idx & 15) << 2;
      float4 v = *(const float4*)&B[(size_t)(k0 + r) * H + n0 + c4];
      bf16 h0 = f2bf(v.x), h1 = f2bf(v.y), h2 = f2bf(v.z), h3 = f2bf(v.w);
      Bth[c4 + 0][r] = h0; Btl[c4 + 0][r] = f2bf(v.x - bfbits2f(__builtin_bit_cast(unsigned short, h0)));
      Bth[c4 + 1][r] = h1; Btl[c4 + 1][r] = f2bf(v.y - bfbits2f(__builtin_bit_cast(unsigned short, h1)));
      Bth[c4 + 2][r] = h2; Btl[c4 + 2][r] = f2bf(v.z - bfbits2f(__builtin_bit_cast(unsigned short, h2)));
      Bth[c4 + 3][r] = h3; Btl[c4 + 3][r] = f2bf(v.w - bfbits2f(__builtin_bit_cast(unsigned short, h3)));
    }
    __syncthreads();

    // ---- fragments: 2 M-tiles x 2 N-tiles per wave, vector LDS loads
    v16bf ah0 = load_frag(&Ah[(wr << 5) + hl][0], g0);
    v16bf al0 = load_frag(&Al[(wr << 5) + hl][0], g0);
    v16bf ah1 = load_frag(&Ah[(wr << 5) + 16 + hl][0], g0);
    v16bf al1 = load_frag(&Al[(wr << 5) + 16 + hl][0], g0);
    v16bf bh0 = load_frag(&Bth[(wc << 5) + hl][0], g0);
    v16bf bl0 = load_frag(&Btl[(wc << 5) + hl][0], g0);
    v16bf bh1 = load_frag(&Bth[(wc << 5) + 16 + hl][0], g0);
    v16bf bl1 = load_frag(&Btl[(wc << 5) + 16 + hl][0], g0);

    // error-compensated bf16x3 per tile
    acc00 = WMMA_BF16(ah0, bh0, acc00);
    acc00 = WMMA_BF16(al0, bh0, acc00);
    acc00 = WMMA_BF16(ah0, bl0, acc00);
    acc01 = WMMA_BF16(ah0, bh1, acc01);
    acc01 = WMMA_BF16(al0, bh1, acc01);
    acc01 = WMMA_BF16(ah0, bl1, acc01);
    acc10 = WMMA_BF16(ah1, bh0, acc10);
    acc10 = WMMA_BF16(al1, bh0, acc10);
    acc10 = WMMA_BF16(ah1, bl0, acc10);
    acc11 = WMMA_BF16(ah1, bh1, acc11);
    acc11 = WMMA_BF16(al1, bh1, acc11);
    acc11 = WMMA_BF16(ah1, bl1, acc11);
  }

  // ---- epilogue: C/D layout: elem r -> row (sm + r + hh*8), col (sn + hl)
  const int col0 = n0 + (wc << 5) + hl;
  const int col1 = col0 + 16;
  const float b0 = bias ? bias[col0] : 0.f;
  const float b1v = bias ? bias[col1] : 0.f;
  const int rbase0 = m0 + (wr << 5) + (hh << 3);
  const int rbase1 = rbase0 + 16;
#pragma unroll
  for (int r = 0; r < 8; ++r) {
    int g0r = rbase0 + r;
    if (g0r < M) {
      C[(size_t)g0r * H + col0] = acc00[r] + b0;
      C[(size_t)g0r * H + col1] = acc01[r] + b1v;
    }
    int g1r = rbase1 + r;
    if (g1r < M) {
      C[(size_t)g1r * H + col0] = acc10[r] + b0;
      C[(size_t)g1r * H + col1] = acc11[r] + b1v;
    }
  }
}

// ---------- small kernels ----------
__global__ void k_zero(float* p, int n) {
  int i = blockIdx.x * blockDim.x + threadIdx.x;
  if (i < n) p[i] = 0.f;
}

__global__ void k_colsum(const float* __restrict__ x, float* __restrict__ g, int N) {
  int c = threadIdx.x;
  int r0 = blockIdx.x * 256;
  int r1 = r0 + 256; if (r1 > N) r1 = N;
  float acc = 0.f;
  for (int r = r0; r < r1; ++r) acc += x[(size_t)r * H + c];
  atomicAdd(&g[c], acc);
}

__global__ void k_selector(const float* __restrict__ g,
                           const float* __restrict__ W1, const float* __restrict__ b1,
                           const float* __restrict__ W2, const float* __restrict__ b2,
                           float* __restrict__ sw, int N) {
  __shared__ float gsh[H];
  __shared__ float hid[128];
  __shared__ float lg[4];
  int t = threadIdx.x;  // 128 threads
  float invN = 1.f / (float)N;
  gsh[t] = g[t] * invN;
  gsh[t + 128] = g[t + 128] * invN;
  __syncthreads();
  float a = b1[t];
  for (int k = 0; k < H; ++k) a += gsh[k] * W1[k * 128 + t];
  hid[t] = fmaxf(a, 0.f);
  __syncthreads();
  if (t < 4) {
    float l = b2[t];
    for (int k = 0; k < 128; ++k) l += hid[k] * W2[k * 4 + t];
    lg[t] = l;
  }
  __syncthreads();
  if (t == 0) {
    float m = fmaxf(fmaxf(lg[0], lg[1]), fmaxf(lg[2], lg[3]));
    float e0 = __expf(lg[0] - m), e1 = __expf(lg[1] - m);
    float e2 = __expf(lg[2] - m), e3 = __expf(lg[3] - m);
    float s = e0 + e1 + e2 + e3;
    sw[0] = e0 / s; sw[1] = e1 / s; sw[2] = e2 / s; sw[3] = e3 / s;
  }
}

// per-node alpha_src / alpha_dst dots (one wave32 per node)
__global__ void k_sd(const float* __restrict__ h, const float* __restrict__ asrc,
                     const float* __restrict__ adst, float* __restrict__ s,
                     float* __restrict__ d, int N) {
  int n = blockIdx.x * 8 + (threadIdx.x >> 5);
  if (n >= N) return;
  int lane = threadIdx.x & 31;
  float vs = 0.f, vd = 0.f;
  for (int c = lane; c < H; c += 32) {
    float v = h[(size_t)n * H + c];
    vs += v * asrc[c];
    vd += v * adst[c];
  }
  for (int o = 16; o; o >>= 1) { vs += __shfl_xor(vs, o, 32); vd += __shfl_xor(vd, o, 32); }
  if (lane == 0) { s[n] = vs; d[n] = vd; }
}

__global__ void k_init_m(const float* __restrict__ s, const float* __restrict__ d,
                         unsigned* __restrict__ mb, int N) {
  int n = blockIdx.x * blockDim.x + threadIdx.x;
  if (n < N) mb[n] = f2ord(lrelu(s[n] + d[n]));   // self-loop logit seeds the max
}

__global__ void k_edge_max(const int* __restrict__ src, const int* __restrict__ dst,
                           const int* __restrict__ attr, int branch,
                           const float* __restrict__ s, const float* __restrict__ d,
                           unsigned* __restrict__ mb, int E) {
  int e = blockIdx.x * blockDim.x + threadIdx.x;
  if (e < E && attr[e] == branch) {
    int dn = dst[e];
    float v = lrelu(s[src[e]] + d[dn]);
    atomicMax(&mb[dn], f2ord(v));
  }
}

// one wave32 per node: w_self, denom init, num init with w_self*h
__global__ void k_init_num(const float* __restrict__ h, const float* __restrict__ s,
                           const float* __restrict__ d, const unsigned* __restrict__ mb,
                           float* __restrict__ mf, float* __restrict__ denom,
                           float* __restrict__ num, int N) {
  int n = blockIdx.x * 8 + (threadIdx.x >> 5);
  if (n >= N) return;
  int lane = threadIdx.x & 31;
  float m = ord2f(mb[n]);
  float wself = __expf(lrelu(s[n] + d[n]) - m);
  if (lane == 0) { mf[n] = m; denom[n] = wself; }
  for (int c = lane; c < H; c += 32) num[(size_t)n * H + c] = wself * h[(size_t)n * H + c];
}

// one wave32 per edge: scatter-add w and w*h[src]
__global__ void k_edge_accum(const int* __restrict__ src, const int* __restrict__ dst,
                             const int* __restrict__ attr, int branch,
                             const float* __restrict__ s, const float* __restrict__ d,
                             const float* __restrict__ mf, const float* __restrict__ h,
                             float* __restrict__ denom, float* __restrict__ num, int E) {
  int e = blockIdx.x * 8 + (threadIdx.x >> 5);
  if (e >= E || attr[e] != branch) return;
  int sn = src[e], dn = dst[e];
  float w = __expf(lrelu(s[sn] + d[dn]) - mf[dn]);
  int lane = threadIdx.x & 31;
  if (lane == 0) atomicAdd(&denom[dn], w);
  for (int c = lane; c < H; c += 32)
    atomicAdd(&num[(size_t)dn * H + c], w * h[(size_t)sn * H + c]);
}

__global__ void k_finalize(float* __restrict__ num, const float* __restrict__ denom,
                           const float* __restrict__ bias, int total) {
  int i = blockIdx.x * blockDim.x + threadIdx.x;
  if (i < total) {
    int n = i >> 8;
    num[i] = num[i] / denom[n] + bias[i & (H - 1)];
  }
}

// LayerNorm + ReLU (+optional softmax-weight scale) per row; one wave32 per node
__global__ void k_ln_relu(const float* __restrict__ t, const float* __restrict__ gamma,
                          const float* __restrict__ beta, float* __restrict__ out,
                          const float* __restrict__ swp, int swidx, int accumulate, int N) {
  int n = blockIdx.x * 8 + (threadIdx.x >> 5);
  if (n >= N) return;
  int lane = threadIdx.x & 31;
  float v[8];
  float sum = 0.f, sq = 0.f;
#pragma unroll
  for (int j = 0; j < 8; ++j) {
    v[j] = t[(size_t)n * H + lane + j * 32];
    sum += v[j];
    sq += v[j] * v[j];
  }
  for (int o = 16; o; o >>= 1) { sum += __shfl_xor(sum, o, 32); sq += __shfl_xor(sq, o, 32); }
  float mu = sum * (1.f / H);
  float var = sq * (1.f / H) - mu * mu;
  float inv = rsqrtf(var + 1e-5f);
  float scale = swp ? swp[swidx] : 1.f;
#pragma unroll
  for (int j = 0; j < 8; ++j) {
    int c = lane + j * 32;
    float y = (v[j] - mu) * inv * gamma[c] + beta[c];
    y = fmaxf(y, 0.f) * scale;
    if (accumulate) out[(size_t)n * H + c] += y;
    else            out[(size_t)n * H + c] = y;
  }
}

// ---------- launcher ----------
extern "C" void kernel_launch(void* const* d_in, const int* in_sizes, int n_in,
                              void* d_out, int out_size, void* d_ws, size_t ws_size,
                              hipStream_t stream) {
  const float* x     = (const float*)d_in[0];
  const int*   ei    = (const int*)d_in[1];   // [2,E]
  const int*   ea    = (const int*)d_in[2];   // [E]
  const float* Wg    = (const float*)d_in[3];
  const float* a_src = (const float*)d_in[4];
  const float* a_dst = (const float*)d_in[5];
  const float* bg    = (const float*)d_in[6];
  const float* Wsl   = (const float*)d_in[7];
  const float* bsl   = (const float*)d_in[8];
  const float* gsl   = (const float*)d_in[9];
  const float* betas = (const float*)d_in[10];
  const float* W1    = (const float*)d_in[11];
  const float* b1    = (const float*)d_in[12];
  const float* W2    = (const float*)d_in[13];
  const float* b2    = (const float*)d_in[14];
  const float* Wf    = (const float*)d_in[15];
  const float* bfv   = (const float*)d_in[16];
  const float* gf    = (const float*)d_in[17];
  const float* betaf = (const float*)d_in[18];

  const int N = in_sizes[0] / H;
  const int E = in_sizes[2];
  float* out = (float*)d_out;

  // workspace carve-out
  char* wsb = (char*)d_ws;
  size_t off = 0;
  auto alloc = [&](size_t bytes) -> void* {
    void* p = wsb + off;
    off += (bytes + 255) & ~(size_t)255;
    return p;
  };
  float*    g     = (float*)alloc(H * sizeof(float));
  float*    sw    = (float*)alloc(4 * sizeof(float));
  float*    sArr  = (float*)alloc((size_t)N * sizeof(float));
  float*    dArr  = (float*)alloc((size_t)N * sizeof(float));
  unsigned* mb    = (unsigned*)alloc((size_t)N * sizeof(unsigned));
  float*    mf    = (float*)alloc((size_t)N * sizeof(float));
  float*    denom = (float*)alloc((size_t)N * sizeof(float));
  float*    bufH  = (float*)alloc((size_t)N * H * sizeof(float));  // h / t
  float*    bufX  = (float*)alloc((size_t)N * H * sizeof(float));  // num / xi
  float*    bufC  = (float*)alloc((size_t)N * H * sizeof(float));  // combined

  dim3 gemmGrid((unsigned)((N + TM - 1) / TM), (unsigned)(H / TNB));
  int nWaveBlocks = (N + 7) / 8;
  int eWaveBlocks = (E + 7) / 8;
  int nThrBlocks  = (N + 255) / 256;
  int eThrBlocks  = (E + 255) / 256;
  int nhBlocks    = (N * H + 255) / 256;

  // strategy weights
  k_zero<<<1, 256, 0, stream>>>(g, H);
  k_colsum<<<nThrBlocks, 256, 0, stream>>>(x, g, N);
  k_selector<<<1, 128, 0, stream>>>(g, W1, b1, W2, b2, sw, N);

  for (int i = 0; i < 3; ++i) {
    k_gemm_bf16x3<<<gemmGrid, 256, 0, stream>>>(x, Wg + (size_t)i * H * H, nullptr, bufH, N);
    k_sd<<<nWaveBlocks, 256, 0, stream>>>(bufH, a_src + i * H, a_dst + i * H, sArr, dArr, N);
    k_init_m<<<nThrBlocks, 256, 0, stream>>>(sArr, dArr, mb, N);
    k_edge_max<<<eThrBlocks, 256, 0, stream>>>(ei, ei + E, ea, i, sArr, dArr, mb, E);
    k_init_num<<<nWaveBlocks, 256, 0, stream>>>(bufH, sArr, dArr, mb, mf, denom, bufX, N);
    k_edge_accum<<<eWaveBlocks, 256, 0, stream>>>(ei, ei + E, ea, i, sArr, dArr, mf, bufH, denom, bufX, E);
    k_finalize<<<nhBlocks, 256, 0, stream>>>(bufX, denom, bg + i * H, N * H);
    k_gemm_bf16x3<<<gemmGrid, 256, 0, stream>>>(bufX, Wsl + (size_t)i * H * H, bsl + i * H, bufH, N);
    k_ln_relu<<<nWaveBlocks, 256, 0, stream>>>(bufH, gsl + i * H, betas + i * H, bufC, sw, i,
                                               i > 0 ? 1 : 0, N);
  }
  // branch 3: identity features
  k_gemm_bf16x3<<<gemmGrid, 256, 0, stream>>>(x, Wsl + (size_t)3 * H * H, bsl + 3 * H, bufH, N);
  k_ln_relu<<<nWaveBlocks, 256, 0, stream>>>(bufH, gsl + 3 * H, betas + 3 * H, bufC, sw, 3, 1, N);

  // fusion block
  k_gemm_bf16x3<<<gemmGrid, 256, 0, stream>>>(bufC, Wf, bfv, bufH, N);
  k_ln_relu<<<nWaveBlocks, 256, 0, stream>>>(bufH, gf, betaf, out, nullptr, 0, 0, N);
}
